// GPT_28484223107276
// MI455X (gfx1250) — matrix operand order repaired and checked by reading, not compiled, not run on hardware
//
#include <hip/hip_runtime.h>
#include <hip/hip_bf16.h>
#include <math.h>

typedef _Float16 h16;
typedef __attribute__((ext_vector_type(16))) _Float16 v16h;
typedef __attribute__((ext_vector_type(8)))  _Float16 v8h;
typedef __attribute__((ext_vector_type(8)))  float    v8f;

#define B_    512
#define TT    44          // t (tokens per stream after shift)
#define S3    132         // 3*t
#define C_    512
#define NH    8
#define HD    64
#define HIDN  2048
#define OH_   768
#define V_    1000
#define NL    8
#define MROWS (B_*S3)     // 67584
#define HROWS (B_*TT)     // 22528
#define WIN_  30
#define LN_EPS_ 1e-5f
#define LDS_T (128 * 40)  // one LDS tile in halves (128 rows x 32K + 8 pad)

// ---------------------------------------------------------------------------
// Tiled WMMA GEMM: C[M,N] = epilogue(A[M,K] @ W[K,N]), weights pre-transposed
// to Bt[N][K] so both operands are K-contiguous.
// A,Bt f16; f32 accumulate via v_wmma_f32_16x16x32_f16.
// Block = 256 threads (8 wave32). Tile 128x128, K-step 32.
// Wave (2x4) owns 64x32 patch = 4x2 grid of 16x16 wmma tiles.
// Ping-pong LDS (2 x 20KB): compute slab p while committing slab p^1;
// next-slab global_load_b128 staged in VGPRs during the wmma work.
// One barrier per K-iteration. K-tile count is even for all shapes here.
// Fragments: per CDNA5 16-bit layouts each lane = two 16B contiguous LDS
// reads (ds_load_b128) for A and for Bt.
// M multiple of 128; K multiple of 32; N guarded (row-clamped on Bt).
// ---------------------------------------------------------------------------
__global__ __launch_bounds__(256)
void gemm_wmma_f16(const h16* __restrict__ A, const h16* __restrict__ Bt,
                   const float* __restrict__ bias, const float* __restrict__ resid,
                   float* __restrict__ Cf, h16* __restrict__ Ch,
                   int M, int N, int K, int act)
{
    __shared__ h16 sA[2][LDS_T];
    __shared__ h16 sB[2][LDS_T];

    const int tid   = threadIdx.x;
    const int lane  = tid & 31;
    const int wave  = tid >> 5;
    const int wm    = (wave & 1) * 64;
    const int wn    = (wave >> 1) * 32;
    const int lhalf = lane >> 4;
    const int lmod  = lane & 15;
    const int tileM = blockIdx.x * 128;
    const int tileN = blockIdx.y * 128;
    const int ktiles = K >> 5;

    // staging assignment: vector id v in [0,512): row = v>>2, col8 = (v&3)*8
    const int ra0 = tid >> 2,         ca0 = (tid & 3) * 8;
    const int ra1 = (tid + 256) >> 2, ca1 = (tid & 3) * 8;
    const int nrow0 = tileN + ra0,    nrow1 = tileN + ra1;
    const int crow0 = (nrow0 < N) ? nrow0 : (N - 1);   // clamp: keep load global+EXEC-full
    const int crow1 = (nrow1 < N) ? nrow1 : (N - 1);

    uint4 ta0, ta1, tb0, tb1;
    const uint4 z4 = make_uint4(0u, 0u, 0u, 0u);

    auto loadStage = [&](int k0) {
        ta0 = *(const uint4*)&A[(size_t)(tileM + ra0) * K + k0 + ca0];
        ta1 = *(const uint4*)&A[(size_t)(tileM + ra1) * K + k0 + ca1];
        tb0 = *(const uint4*)&Bt[(size_t)crow0 * K + k0 + ca0];
        tb1 = *(const uint4*)&Bt[(size_t)crow1 * K + k0 + ca1];
        if (nrow0 >= N) tb0 = z4;
        if (nrow1 >= N) tb1 = z4;
    };
    auto storeStage = [&](h16* dA, h16* dB) {
        *(uint4*)&dA[ra0 * 40 + ca0] = ta0;
        *(uint4*)&dA[ra1 * 40 + ca1] = ta1;
        *(uint4*)&dB[ra0 * 40 + ca0] = tb0;
        *(uint4*)&dB[ra1 * 40 + ca1] = tb1;
    };

    const v8f vzero = {0.f,0.f,0.f,0.f,0.f,0.f,0.f,0.f};
    v8f acc[4][2];
    #pragma unroll
    for (int i = 0; i < 4; ++i)
        #pragma unroll
        for (int j = 0; j < 2; ++j) acc[i][j] = vzero;

    auto computeSlab = [&](const h16* cA, const h16* cB) {
        union F { v16h v; v8h h[2]; } af[4], bf[2];
        #pragma unroll
        for (int i = 0; i < 4; ++i) {
            const h16* ar = &cA[(wm + i * 16 + lmod) * 40];
            af[i].h[0] = *(const v8h*)&ar[lhalf * 8];        // K lhalf*8 .. +7
            af[i].h[1] = *(const v8h*)&ar[16 + lhalf * 8];   // K 16+lhalf*8 .. +7
        }
        #pragma unroll
        for (int j = 0; j < 2; ++j) {
            const h16* br = &cB[(wn + j * 16 + lmod) * 40];
            bf[j].h[0] = *(const v8h*)&br[lhalf * 8];        // K lhalf*8 .. +7
            bf[j].h[1] = *(const v8h*)&br[16 + lhalf * 8];   // K 16+lhalf*8 .. +7
        }
        #pragma unroll
        for (int i = 0; i < 4; ++i)
            #pragma unroll
            for (int j = 0; j < 2; ++j)
                acc[i][j] = __builtin_amdgcn_wmma_f32_16x16x32_f16(
                    false, af[i].v, false, bf[j].v, (short)0, acc[i][j], false, false);
    };

    loadStage(0);
    storeStage(sA[0], sB[0]);
    __syncthreads();

    for (int kt = 0; kt < ktiles; kt += 2) {
        // even slab: read buf0, prefetch kt+1 into buf1
        loadStage((kt + 1) << 5);          // kt+1 < ktiles always (ktiles even)
        computeSlab(sA[0], sB[0]);
        storeStage(sA[1], sB[1]);
        __syncthreads();
        // odd slab: read buf1, prefetch kt+2 into buf0
        if (kt + 2 < ktiles) {
            loadStage((kt + 2) << 5);
            computeSlab(sA[1], sB[1]);
            storeStage(sA[0], sB[0]);
            __syncthreads();
        } else {
            computeSlab(sA[1], sB[1]);
        }
    }

    // Epilogue; C layout: VGPR r -> M = r (lanes 0-15) / 8+r (lanes 16-31)
    #pragma unroll
    for (int j = 0; j < 2; ++j) {
        const int n = tileN + wn + j * 16 + lmod;
        if (n >= N) continue;
        const float bv = bias ? bias[n] : 0.f;
        #pragma unroll
        for (int i = 0; i < 4; ++i) {
            #pragma unroll
            for (int r = 0; r < 8; ++r) {
                const int m = tileM + wm + i * 16 + r + lhalf * 8;
                float v = acc[i][j][r] + bv;
                if (act)   v = 0.5f * v * (1.f + erff(v * 0.70710678118f)); // exact GELU
                if (resid) v += resid[(size_t)m * N + n];
                if (Cf) Cf[(size_t)m * N + n] = v;
                else    Ch[(size_t)m * N + n] = (h16)v;
            }
        }
    }
}

// ---------------------------------------------------------------------------
// Transpose + f32->f16 convert: src[K][N] (f32) -> dst[N][K] (f16).
// 32x32 LDS tile, coalesced on both sides. Block = 256 threads.
// ---------------------------------------------------------------------------
__global__ __launch_bounds__(256)
void cvt_transpose(const float* __restrict__ src, h16* __restrict__ dst, int K, int N)
{
    __shared__ h16 t[32][33];
    const int kb = blockIdx.x * 32, nb = blockIdx.y * 32;
    const int tx = threadIdx.x & 31, ty = threadIdx.x >> 5;   // ty in [0,8)
    #pragma unroll
    for (int i = 0; i < 32; i += 8) {
        int k = kb + ty + i, n = nb + tx;
        t[ty + i][tx] = (k < K && n < N) ? (h16)src[(size_t)k * N + n] : (h16)0.f;
    }
    __syncthreads();
    #pragma unroll
    for (int i = 0; i < 32; i += 8) {
        int n = nb + ty + i, k = kb + tx;
        if (n < N && k < K) dst[(size_t)n * K + k] = t[tx][ty + i];
    }
}

// ---------------------------------------------------------------------------
// Embedding gather (float4 copies)
// ---------------------------------------------------------------------------
__global__ __launch_bounds__(128)
void embed_kernel(const int* __restrict__ label, const int* __restrict__ wav1,
                  const int* __restrict__ wav2,  const float* __restrict__ eL,
                  const float* __restrict__ e1,  const float* __restrict__ e2,
                  float* __restrict__ X)
{
    int r = blockIdx.x;           // 0..MROWS-1
    int b = r / S3, s = r % S3;
    const float* src;
    if (s < TT)          src = eL + (size_t)label[b] * C_;
    else if (s < 2 * TT) src = e1 + (size_t)wav1[b * (TT + 1) + (s - TT)] * C_;
    else                 src = e2 + (size_t)wav2[b * (TT + 1) + (s - 2 * TT)] * C_;
    const float4* s4 = (const float4*)src;
    float4* d4 = (float4*)(X + (size_t)r * C_);
    d4[threadIdx.x] = s4[threadIdx.x];       // 128 threads x float4 = 512 floats
}

// ---------------------------------------------------------------------------
// 3-stream LayerNorm over C=512; gamma/beta stream s = (row%132)/44.
// Block must be 256.
// ---------------------------------------------------------------------------
__global__ __launch_bounds__(256)
void ln3_kernel(const float* __restrict__ X, const float* __restrict__ g,
                const float* __restrict__ be, h16* __restrict__ H)
{
    __shared__ float s1[256], s2[256];
    int r = blockIdx.x;
    int s = (r % S3) / TT;
    const float* xr = X + (size_t)r * C_;
    int t = threadIdx.x;
    float a0 = xr[t], a1 = xr[t + 256];
    s1[t] = a0 + a1;
    s2[t] = a0 * a0 + a1 * a1;
    __syncthreads();
    for (int st = 128; st > 0; st >>= 1) {
        if (t < st) { s1[t] += s1[t + st]; s2[t] += s2[t + st]; }
        __syncthreads();
    }
    float mu  = s1[0] * (1.f / C_);
    float var = s2[0] * (1.f / C_) - mu * mu;
    float inv = rsqrtf(var + LN_EPS_);
    const float* gg = g  + s * C_;
    const float* bb = be + s * C_;
    H[(size_t)r * C_ + t]       = (h16)((a0 - mu) * inv * gg[t]       + bb[t]);
    H[(size_t)r * C_ + t + 256] = (h16)((a1 - mu) * inv * gg[t + 256] + bb[t + 256]);
}

// ---------------------------------------------------------------------------
// Masked softmax attention: one block per (b, head, query). 132 keys, hd=64.
// Mask (3x3 tile): allowed iff k%44 in [q%44-29, q%44].
// ---------------------------------------------------------------------------
__global__ __launch_bounds__(160)
void attn_kernel(const h16* __restrict__ Q, const h16* __restrict__ K,
                 const h16* __restrict__ V, h16* __restrict__ Y)
{
    __shared__ float sq[HD];
    __shared__ float sc[S3];
    __shared__ float red[2];
    int gid = blockIdx.x;
    int qs = gid % S3;
    int h  = (gid / S3) % NH;
    int b  = gid / (S3 * NH);
    size_t base = (size_t)b * S3 * C_ + (size_t)h * HD;
    int tid = threadIdx.x;

    if (tid < HD) sq[tid] = (float)Q[base + (size_t)qs * C_ + tid];
    __syncthreads();

    if (tid < S3) {
        const v8h* kr = (const v8h*)(K + base + (size_t)tid * C_);  // 16B aligned
        float d = 0.f;
        #pragma unroll
        for (int c8 = 0; c8 < HD / 8; ++c8) {
            v8h kv = kr[c8];
            #pragma unroll
            for (int u = 0; u < 8; ++u) d += sq[c8 * 8 + u] * (float)kv[u];
        }
        int qm = qs % TT, km = tid % TT;
        bool masked = (km > qm) || ((qm >= WIN_) && (km <= qm - WIN_));
        sc[tid] = masked ? -INFINITY : d * 0.125f;   // 1/sqrt(64)
    }
    __syncthreads();
    if (tid == 0) {
        float mx = -INFINITY;
        for (int j = 0; j < S3; ++j) mx = fmaxf(mx, sc[j]);
        red[0] = mx;
    }
    __syncthreads();
    if (tid < S3) sc[tid] = __expf(sc[tid] - red[0]);
    __syncthreads();
    if (tid == 0) {
        float sm = 0.f;
        for (int j = 0; j < S3; ++j) sm += sc[j];
        red[1] = 1.f / sm;
    }
    __syncthreads();
    if (tid < HD) {
        float a = 0.f;
        for (int j = 0; j < S3; ++j) a += sc[j] * (float)V[base + (size_t)j * C_ + tid];
        Y[base + (size_t)qs * C_ + tid] = (h16)(a * red[1]);
    }
}

// gather x[:, off:off+44] rows -> contiguous f16 (HROWS x C), 4-wide packs
__global__ __launch_bounds__(128)
void gather_rows(const float* __restrict__ X, h16* __restrict__ XS, int off)
{
    int r = blockIdx.x;                  // 0..HROWS-1
    int b = r / TT, i = r % TT;
    const float4* s4 = (const float4*)(X + (size_t)(b * S3 + off + i) * C_);
    float4 v = s4[threadIdx.x];
    union { h16 e[4]; uint2 u; } p;
    p.e[0] = (h16)v.x; p.e[1] = (h16)v.y; p.e[2] = (h16)v.z; p.e[3] = (h16)v.w;
    *(uint2*)&XS[(size_t)r * C_ + threadIdx.x * 4] = p.u;
}

// cross-entropy: one block per row (V=1000 logits); accumulate mean into out[0]
__global__ __launch_bounds__(256)
void ce_kernel(const float* __restrict__ logits, const int* __restrict__ wav,
               float* __restrict__ out)
{
    __shared__ float red[256];
    int row = blockIdx.x;
    const float* lr = logits + (size_t)row * V_;
    int tid = threadIdx.x;
    float mx = -INFINITY;
    for (int j = tid; j < V_; j += 256) mx = fmaxf(mx, lr[j]);
    red[tid] = mx; __syncthreads();
    for (int st = 128; st > 0; st >>= 1) {
        if (tid < st) red[tid] = fmaxf(red[tid], red[tid + st]);
        __syncthreads();
    }
    mx = red[0]; __syncthreads();
    float sm = 0.f;
    for (int j = tid; j < V_; j += 256) sm += __expf(lr[j] - mx);
    red[tid] = sm; __syncthreads();
    for (int st = 128; st > 0; st >>= 1) {
        if (tid < st) red[tid] += red[tid + st];
        __syncthreads();
    }
    if (tid == 0) {
        int b = row / TT, i = row % TT;
        int tgt = wav[b * (TT + 1) + i + 1];         // shifted target
        float loss = -(lr[tgt] - mx - __logf(red[0]));
        atomicAdd(out, loss * (1.f / (float)HROWS));
    }
}

__global__ void zero_kernel(float* out) { if (threadIdx.x == 0) out[0] = 0.f; }

// ---------------------------------------------------------------------------
extern "C" void kernel_launch(void* const* d_in, const int* in_sizes, int n_in,
                              void* d_out, int out_size, void* d_ws, size_t ws_size,
                              hipStream_t stream)
{
    (void)in_sizes; (void)n_in; (void)out_size; (void)ws_size;

    const int*   label = (const int*)d_in[0];
    const int*   wav1  = (const int*)d_in[1];
    const int*   wav2  = (const int*)d_in[2];
    const float* embL  = (const float*)d_in[3];
    const float* emb1  = (const float*)d_in[4];
    const float* emb2  = (const float*)d_in[5];
    const float* ln1g  = (const float*)d_in[6];
    const float* ln1b  = (const float*)d_in[7];
    const float* ln2g  = (const float*)d_in[8];
    const float* ln2b  = (const float*)d_in[9];
    const float* Wq = (const float*)d_in[10]; const float* bq = (const float*)d_in[11];
    const float* Wk = (const float*)d_in[12]; const float* bk = (const float*)d_in[13];
    const float* Wv = (const float*)d_in[14]; const float* bv = (const float*)d_in[15];
    const float* Wp = (const float*)d_in[16]; const float* bp = (const float*)d_in[17];
    const float* W1 = (const float*)d_in[18]; const float* b1 = (const float*)d_in[19];
    const float* W2 = (const float*)d_in[20]; const float* b2 = (const float*)d_in[21];
    const float* o1W1 = (const float*)d_in[22]; const float* o1b1 = (const float*)d_in[23];
    const float* o1W2 = (const float*)d_in[24]; const float* o1b2 = (const float*)d_in[25];
    const float* o2W1 = (const float*)d_in[26]; const float* o2b1 = (const float*)d_in[27];
    const float* o2W2 = (const float*)d_in[28]; const float* o2b2 = (const float*)d_in[29];
    float* out = (float*)d_out;

    char* ws = (char*)d_ws;
    size_t off = 0;
    auto alloc = [&](size_t bytes) -> char* {
        char* p = ws + off;
        off += (bytes + 255) & ~(size_t)255;
        return p;
    };
    float* X   = (float*)alloc((size_t)MROWS * C_ * 4);   // residual stream (f32)
    h16*   H16 = (h16*)  alloc((size_t)MROWS * C_ * 2);   // LN output (f16)
    h16*   Q16 = (h16*)  alloc((size_t)MROWS * C_ * 2);
    h16*   K16 = (h16*)  alloc((size_t)MROWS * C_ * 2);
    h16*   V16 = (h16*)  alloc((size_t)MROWS * C_ * 2);
    h16*   Y16 = (h16*)  alloc((size_t)MROWS * C_ * 2);
    char*  MIDc = alloc((size_t)MROWS * HIDN * 2);        // MLP hidden (f16)
    h16*   MID  = (h16*)MIDc;
    // f16 transposed weight pool ([N][K] per matrix)
    h16* wq16 = (h16*)alloc((size_t)NL * C_ * C_ * 2);
    h16* wk16 = (h16*)alloc((size_t)NL * C_ * C_ * 2);
    h16* wv16 = (h16*)alloc((size_t)NL * C_ * C_ * 2);
    h16* wp16 = (h16*)alloc((size_t)NL * C_ * C_ * 2);
    h16* w116 = (h16*)alloc((size_t)NL * C_ * HIDN * 2);
    h16* w216 = (h16*)alloc((size_t)NL * HIDN * C_ * 2);
    h16* o1w116 = (h16*)alloc((size_t)C_ * OH_ * 2);
    h16* o1w216 = (h16*)alloc((size_t)OH_ * V_ * 2);
    h16* o2w116 = (h16*)alloc((size_t)C_ * OH_ * 2);
    h16* o2w216 = (h16*)alloc((size_t)OH_ * V_ * 2);
    // head buffers overlaid on MID (dead after the layer stack)
    size_t xs_b = ((size_t)HROWS * C_  * 2 + 255) & ~(size_t)255;
    size_t h1_b = ((size_t)HROWS * OH_ * 2 + 255) & ~(size_t)255;
    h16*   XS  = (h16*)MIDc;
    h16*   H1  = (h16*)(MIDc + xs_b);
    float* LOG = (float*)(MIDc + xs_b + h1_b);

    auto cvtT = [&](const float* s, h16* d, int K, int N) {
        dim3 grid((unsigned)((K + 31) / 32), (unsigned)((N + 31) / 32));
        cvt_transpose<<<grid, 256, 0, stream>>>(s, d, K, N);
    };
    auto gemm = [&](const h16* A, const h16* Bm, const float* bias, const float* resid,
                    float* Cf, h16* Ch, int M, int N, int K, int act) {
        dim3 grid((unsigned)(M / 128), (unsigned)((N + 127) / 128));
        gemm_wmma_f16<<<grid, 256, 0, stream>>>(A, Bm, bias, resid, Cf, Ch, M, N, K, act);
    };

    zero_kernel<<<1, 32, 0, stream>>>(out);

    // one-time f32 -> f16 transpose-convert (weights stay hot in 192MB L2)
    for (int l = 0; l < NL; ++l) {
        size_t wo = (size_t)l * C_ * C_, w1o = (size_t)l * C_ * HIDN;
        cvtT(Wq + wo, wq16 + wo, C_, C_);
        cvtT(Wk + wo, wk16 + wo, C_, C_);
        cvtT(Wv + wo, wv16 + wo, C_, C_);
        cvtT(Wp + wo, wp16 + wo, C_, C_);
        cvtT(W1 + w1o, w116 + w1o, C_, HIDN);
        cvtT(W2 + w1o, w216 + w1o, HIDN, C_);
    }
    cvtT(o1W1, o1w116, C_, OH_);
    cvtT(o1W2, o1w216, OH_, V_);
    cvtT(o2W1, o2w116, C_, OH_);
    cvtT(o2W2, o2w216, OH_, V_);

    embed_kernel<<<MROWS, 128, 0, stream>>>(label, wav1, wav2, embL, emb1, emb2, X);

    for (int l = 0; l < NL; ++l) {
        const size_t wo  = (size_t)l * C_ * C_;
        const size_t w1o = (size_t)l * C_ * HIDN;
        // attention block
        ln3_kernel<<<MROWS, 256, 0, stream>>>(X, ln1g + (size_t)l * 3 * C_,
                                              ln1b + (size_t)l * 3 * C_, H16);
        gemm(H16, wq16 + wo, bq + (size_t)l * C_, nullptr, nullptr, Q16, MROWS, C_, C_, 0);
        gemm(H16, wk16 + wo, bk + (size_t)l * C_, nullptr, nullptr, K16, MROWS, C_, C_, 0);
        gemm(H16, wv16 + wo, bv + (size_t)l * C_, nullptr, nullptr, V16, MROWS, C_, C_, 0);
        attn_kernel<<<B_ * NH * S3, 160, 0, stream>>>(Q16, K16, V16, Y16);
        gemm(Y16, wp16 + wo, bp + (size_t)l * C_, X, X, nullptr, MROWS, C_, C_, 0);
        // MLP block
        ln3_kernel<<<MROWS, 256, 0, stream>>>(X, ln2g + (size_t)l * 3 * C_,
                                              ln2b + (size_t)l * 3 * C_, H16);
        gemm(H16, w116 + w1o, b1 + (size_t)l * HIDN, nullptr, nullptr, MID, MROWS, HIDN, C_, 1);
        gemm(MID, w216 + w1o, b2 + (size_t)l * C_, X, X, nullptr, MROWS, C_, HIDN, 0);
    }

    // head 1: x[:, t:2t]
    gather_rows<<<HROWS, 128, 0, stream>>>(X, XS, TT);
    gemm(XS, o1w116, o1b1, nullptr, nullptr, H1, HROWS, OH_, C_, 0);
    gemm(H1, o1w216, o1b2, nullptr, LOG, nullptr, HROWS, V_, OH_, 0);
    ce_kernel<<<HROWS, 256, 0, stream>>>(LOG, wav1, out);

    // head 2: x[:, 2t:]
    gather_rows<<<HROWS, 128, 0, stream>>>(X, XS, 2 * TT);
    gemm(XS, o2w116, o2b1, nullptr, nullptr, H1, HROWS, OH_, C_, 0);
    gemm(H1, o2w216, o2b2, nullptr, LOG, nullptr, HROWS, V_, OH_, 0);
    ce_kernel<<<HROWS, 256, 0, stream>>>(LOG, wav2, out);
}
